// CausalGNN_10110353015176
// MI455X (gfx1250) — matrix operand order repaired and checked by reading, not compiled, not run on hardware
//
#include <hip/hip_runtime.h>
#include <hip/hip_bf16.h>

#define NNODES 50000
#define NEDGES 800000
#define NB 50
#define FIN 16
#define HIDC 64
#define NH 4
#define NOUT 3
#define ETOT (NEDGES + NNODES)

typedef __attribute__((ext_vector_type(16))) _Float16 v16h;
typedef __attribute__((ext_vector_type(8)))  _Float16 v8h;
typedef __attribute__((ext_vector_type(8)))  float    v8f;

// ---- order-preserving float<->unsigned key for atomicMax on floats ----
__device__ __forceinline__ unsigned fkey(float f) {
  unsigned u = __float_as_uint(f);
  return (u & 0x80000000u) ? ~u : (u | 0x80000000u);
}
__device__ __forceinline__ float funkey(unsigned k) {
  unsigned u = (k & 0x80000000u) ? (k ^ 0x80000000u) : ~k;
  return __uint_as_float(u);
}
#define NEG_INF_KEY 0x007FFFFFu  // fkey(-inf)

__device__ __forceinline__ float eluf(float v) { return v > 0.f ? v : (__expf(v) - 1.f); }

// -------------------- utility fills --------------------
__global__ void k_fill_f32(float* p, float v, int n) {
  int t = blockIdx.x * blockDim.x + threadIdx.x;
  if (t < n) p[t] = v;
}
__global__ void k_fill_f32x4(float4* p, float v, int n4) {
  int t = blockIdx.x * blockDim.x + threadIdx.x;
  if (t < n4) p[t] = make_float4(v, v, v, v);
}
__global__ void k_fill_u32(unsigned* p, unsigned v, int n) {
  int t = blockIdx.x * blockDim.x + threadIdx.x;
  if (t < n) p[t] = v;
}

// -------------------- edge encoder: sigmoid(relu(ea@W1+b1)@W2+b2) --------------------
__global__ void k_edge_encoder(const float* __restrict__ ea, const float* __restrict__ w1,
                               const float* __restrict__ b1, const float* __restrict__ w2,
                               const float* __restrict__ b2, float* __restrict__ ew,
                               float* __restrict__ ewsum, int E_) {
  int e = blockIdx.x * blockDim.x + threadIdx.x;
  float val = 0.f;
  if (e < E_) {
    float a[5];
#pragma unroll
    for (int i = 0; i < 5; ++i) a[i] = ea[(size_t)e * 5 + i];
    float acc = b2[0];
    for (int j = 0; j < 64; ++j) {
      float hsum = b1[j];
#pragma unroll
      for (int i = 0; i < 5; ++i) hsum += a[i] * w1[i * 64 + j];
      hsum = hsum > 0.f ? hsum : 0.f;
      acc += hsum * w2[j];
    }
    float sg = 1.f / (1.f + __expf(-acc));
    ew[e] = sg;
    val = sg;
  }
  __shared__ float red[256];
  red[threadIdx.x] = val;
  __syncthreads();
  for (int off = 128; off > 0; off >>= 1) {
    if ((int)threadIdx.x < off) red[threadIdx.x] += red[threadIdx.x + off];
    __syncthreads();
  }
  if (threadIdx.x == 0) atomicAdd(ewsum, red[0]);
}

__global__ void k_selfloops(float* ew, const float* ewsum, int E_, int Nn) {
  int i = blockIdx.x * blockDim.x + threadIdx.x;
  if (i < Nn) ew[E_ + i] = ewsum[0] * (1.f / (float)E_);
}

// -------------------- f32 -> f16 packers --------------------
// src [M,Kin] f32 row-major -> dst [M,Kpad] f16 (zero padded)  (layer-0 input only)
__global__ void k_pack_a16(const float* __restrict__ src, _Float16* __restrict__ dst,
                           int M, int Kin, int Kpad) {
  int t = blockIdx.x * blockDim.x + threadIdx.x;
  if (t >= M * Kpad) return;
  int r = t / Kpad, k = t - r * Kpad;
  dst[t] = (k < Kin) ? (_Float16)src[(size_t)r * Kin + k] : (_Float16)0.f;
}
// W [K,256] f32 -> BT [256,Kpad] f16 (per-column contiguous, zero padded)
__global__ void k_pack_bt16(const float* __restrict__ W, _Float16* __restrict__ BT,
                            int K, int Kpad) {
  int t = blockIdx.x * blockDim.x + threadIdx.x;
  if (t >= 256 * Kpad) return;
  int n = t / Kpad, k = t - n * Kpad;
  BT[t] = (k < K) ? (_Float16)W[(size_t)k * 256 + n] : (_Float16)0.f;
}

// -------------------- WMMA GEMM: C[M,256] = A[M,K] x B[K,256] --------------------
// A row-major f16; BT = B^T (per-column contiguous) f16, staged in LDS per block.
// Block = 512 threads = 16 waves; wave nt computes output tile (blockIdx.x, nt).
// Fragment layouts per ISA 7.12.2; fully unrolled K (template), loads clause up front.
template <int K>
__global__ void __launch_bounds__(512)
k_gemm_wmma(const _Float16* __restrict__ A, const _Float16* __restrict__ BT,
            float* __restrict__ C, int M) {
  __shared__ _Float16 sBT[256 * K];
  constexpr int TOT8 = 256 * K / 8;  // v8h chunks of BT
#pragma unroll
  for (int i = threadIdx.x; i < TOT8; i += 512)
    *(v8h*)(sBT + (size_t)i * 8) = *(const v8h*)(BT + (size_t)i * 8);
  __syncthreads();

  int mt   = blockIdx.x;
  int nt   = threadIdx.x >> 5;
  int lane = threadIdx.x & 31;
  int m    = lane & 15;
  int hi   = lane >> 4;
  if (mt * 16 >= M) return;

  const _Float16* arow = A + (size_t)(mt * 16 + m) * K;
  const _Float16* bcol = sBT + (nt * 16 + m) * K;
  __builtin_prefetch(arow + (size_t)16 * K, 0, 0);  // next M-tile row (speculative)

  v8f acc = {};
#pragma unroll
  for (int k0 = 0; k0 < K; k0 += 32) {
    // A 16x32 fragment: elems 0..7 -> K=k0+hi*8+0..7 ; elems 8..15 -> +16
    v8h a0 = *(const v8h*)(arow + k0 + hi * 8);
    v8h a1 = *(const v8h*)(arow + k0 + hi * 8 + 16);
    // B 32x16 fragment: elems 0..7 -> K=k0+hi*8+0..7 ; elems 8..15 -> +16
    v8h b0 = *(const v8h*)(bcol + k0 + hi * 8);
    v8h b1 = *(const v8h*)(bcol + k0 + hi * 8 + 16);
    v16h av, bv;
#pragma unroll
    for (int i = 0; i < 8; ++i) {
      av[i] = a0[i]; av[i + 8] = a1[i];
      bv[i] = b0[i]; bv[i + 8] = b1[i];
    }
    acc = __builtin_amdgcn_wmma_f32_16x16x32_f16(false, av, false, bv, (short)0, acc,
                                                 false, false);
  }
  // C/D layout: VGPR j -> row (hi*8 + j), col = lane%16
  int crow = mt * 16 + hi * 8;
  int ccol = nt * 16 + m;
#pragma unroll
  for (int j = 0; j < 8; ++j) C[(size_t)(crow + j) * 256 + ccol] = acc[j];
}

// -------------------- per-node alpha dots --------------------
__global__ void k_alpha_nodes(const float* __restrict__ h, const float* __restrict__ aw_s,
                              const float* __restrict__ aw_d, float* __restrict__ asrc,
                              float* __restrict__ adst, int Nn) {
  int t = blockIdx.x * blockDim.x + threadIdx.x;
  if (t >= Nn * NH) return;
  int n = t >> 2, hd = t & 3;
  const float* hp = h + (size_t)n * 256 + hd * 64;
  float s1 = 0.f, s2 = 0.f;
  for (int c = 0; c < 64; ++c) {
    float v = hp[c];
    s1 += v * aw_s[hd * 64 + c];
    s2 += v * aw_d[hd * 64 + c];
  }
  asrc[t] = s1;
  adst[t] = s2;
}

// ce[hd] = dot(We[hd,:], att_e[hd,:])  (alpha_e = ew * ce[hd])
__global__ void k_ce(const float* __restrict__ We_l, const float* __restrict__ ae_l,
                     float* __restrict__ ce) {
  int hd = threadIdx.x;
  if (hd < NH) {
    float s = 0.f;
    for (int c = 0; c < 64; ++c) s += We_l[hd * 64 + c] * ae_l[hd * 64 + c];
    ce[hd] = s;
  }
}

// -------------------- edge pass 1: leaky-relu alpha + segment max --------------------
__global__ void k_edge_alpha(const int* __restrict__ srcI, const int* __restrict__ dstI,
                             const float* __restrict__ ew, const float* __restrict__ ce,
                             const float* __restrict__ asrc, const float* __restrict__ adst,
                             float* __restrict__ albuf, unsigned* __restrict__ mbuf,
                             int Et_, int E_) {
  int t = blockIdx.x * blockDim.x + threadIdx.x;
  if (t >= Et_ * NH) return;
  int e = t >> 2, hd = t & 3;
  int s, d;
  if (e < E_) { s = srcI[e]; d = dstI[e]; } else { s = e - E_; d = s; }
  float a = asrc[s * NH + hd] + adst[d * NH + hd] + ew[e] * ce[hd];
  a = a > 0.f ? a : 0.2f * a;
  albuf[t] = a;
  atomicMax(&mbuf[d * NH + hd], fkey(a));
}

// -------------------- edge pass 2: exp(alpha - max) + segment sum --------------------
__global__ void k_edge_softmax(const int* __restrict__ dstI, float* __restrict__ albuf,
                               const unsigned* __restrict__ mbuf, float* __restrict__ sbuf,
                               int Et_, int E_) {
  int t = blockIdx.x * blockDim.x + threadIdx.x;
  if (t >= Et_ * NH) return;
  int e = t >> 2, hd = t & 3;
  int d = (e < E_) ? dstI[e] : (e - E_);
  float mv = funkey(mbuf[d * NH + hd]);
  float pp = __expf(albuf[t] - mv);
  albuf[t] = pp;
  atomicAdd(&sbuf[d * NH + hd], pp);
}

// -------------------- edge pass 3: weighted message scatter --------------------
// 256 threads = 4 edges x 64 quads; float4 gather of h[src], 4 scalar f32 atomics.
__global__ void k_aggregate(const int* __restrict__ srcI, const int* __restrict__ dstI,
                            const float* __restrict__ h, const float* __restrict__ albuf,
                            const float* __restrict__ sbuf, float* __restrict__ outacc,
                            int Et_, int E_) {
  int tid  = threadIdx.x;
  int slot = tid >> 6;           // 4 edges in flight per block
  int q    = (tid & 63) * 4;     // channel base (16B aligned)
  int hd   = q >> 6;             // head is uniform within a quad
  int e = blockIdx.x * 4 + slot;
  if (e >= Et_) return;
  int s, d;
  if (e < E_) { s = srcI[e]; d = dstI[e]; } else { s = e - E_; d = s; }
  float a = albuf[(size_t)e * NH + hd] / (sbuf[d * NH + hd] + 1e-16f);
  float4 hv = *(const float4*)(h + (size_t)s * 256 + q);
  float* op = outacc + (size_t)d * 256 + q;
  atomicAdd(op + 0, hv.x * a);
  atomicAdd(op + 1, hv.y * a);
  atomicAdd(op + 2, hv.z * a);
  atomicAdd(op + 3, hv.w * a);
}

// ---------- head mean + bias (+ residual) + ELU, fused f16 repack for next GEMM ----------
__global__ void k_epilogue(const float* __restrict__ outacc, const float* __restrict__ bias_l,
                           float* __restrict__ xi, _Float16* __restrict__ A16,
                           int Nn, int residual) {
  int t = blockIdx.x * blockDim.x + threadIdx.x;
  if (t >= Nn * 64) return;
  int n = t >> 6, c = t & 63;
  const float* op = outacc + (size_t)n * 256;
  float v = 0.25f * (op[c] + op[64 + c] + op[128 + c] + op[192 + c]) + bias_l[c];
  if (residual) v += xi[t];
  float r = eluf(v);
  xi[t]  = r;
  A16[t] = (_Float16)r;   // next layer's GEMM A operand ([N,64] f16)
}

// -------------------- readout: searchsorted + LN + MLP (targets only) --------------------
__global__ void k_readout(const float* __restrict__ xi, const int* __restrict__ batch,
                          const float* __restrict__ ln_g, const float* __restrict__ ln_b,
                          const float* __restrict__ l1w, const float* __restrict__ l1b,
                          const float* __restrict__ l2w, const float* __restrict__ l2b,
                          float* __restrict__ outp, int Nn) {
  __shared__ float row[64], rln[64], tb[64];
  __shared__ float stat[2];
  __shared__ int rowidx;
  int b = blockIdx.x;
  int c = threadIdx.x;
  if (c == 0) {  // lower_bound(batch, b)
    int lo = 0, hi = Nn;
    while (lo < hi) { int mid = (lo + hi) >> 1; if (batch[mid] < b) lo = mid + 1; else hi = mid; }
    rowidx = lo;
  }
  __syncthreads();
  row[c] = xi[(size_t)rowidx * 64 + c];
  __syncthreads();
  if (c == 0) {
    float s = 0.f;
    for (int k = 0; k < 64; ++k) s += row[k];
    float mu = s * (1.f / 64.f);
    float v = 0.f;
    for (int k = 0; k < 64; ++k) { float d2 = row[k] - mu; v += d2 * d2; }
    stat[0] = mu;
    stat[1] = v * (1.f / 64.f);
  }
  __syncthreads();
  rln[c] = (row[c] - stat[0]) * rsqrtf(stat[1] + 1e-5f) * ln_g[c] + ln_b[c];
  __syncthreads();
  float t = l1b[c];
  for (int k = 0; k < 64; ++k) t += rln[k] * l1w[k * 64 + c];
  tb[c] = eluf(t);
  __syncthreads();
  if (c < NOUT) {
    float y = l2b[c];
    for (int k = 0; k < 64; ++k) y += tb[k] * l2w[k * NOUT + c];
    outp[b * NOUT + c] = y;
  }
}

// ==================== host launch ====================
extern "C" void kernel_launch(void* const* d_in, const int* in_sizes, int n_in,
                              void* d_out, int out_size, void* d_ws, size_t ws_size,
                              hipStream_t stream) {
  const float* x      = (const float*)d_in[0];
  const int*   eidx   = (const int*)  d_in[1];
  const float* eattr  = (const float*)d_in[2];
  const int*   batch  = (const int*)  d_in[3];
  const float* ee_w1  = (const float*)d_in[4];
  const float* ee_b1  = (const float*)d_in[5];
  const float* ee_w2  = (const float*)d_in[6];
  const float* ee_b2  = (const float*)d_in[7];
  const float* W1     = (const float*)d_in[8];
  const float* Ws     = (const float*)d_in[9];
  const float* att_s  = (const float*)d_in[10];
  const float* att_d  = (const float*)d_in[11];
  const float* We     = (const float*)d_in[12];
  const float* att_e  = (const float*)d_in[13];
  const float* bias   = (const float*)d_in[14];
  const float* ln_g   = (const float*)d_in[15];
  const float* ln_b   = (const float*)d_in[16];
  const float* lin1_w = (const float*)d_in[17];
  const float* lin1_b = (const float*)d_in[18];
  const float* lin2_w = (const float*)d_in[19];
  const float* lin2_b = (const float*)d_in[20];
  float* outp = (float*)d_out;

  char* ws = (char*)d_ws;
  size_t off = 0;
  auto carve = [&](size_t bytes) -> char* {
    char* p = ws + off;
    off = (off + bytes + 255) & ~(size_t)255;
    return p;
  };
  float*     ewsum  = (float*)    carve(4);
  float*     ce     = (float*)    carve(16);
  float*     ew     = (float*)    carve((size_t)ETOT * 4);
  _Float16*  A16    = (_Float16*) carve((size_t)NNODES * 64 * 2);
  _Float16*  BT16   = (_Float16*) carve((size_t)256 * 64 * 2);
  float*     hbuf   = (float*)    carve((size_t)NNODES * 256 * 4);
  float*     asrc   = (float*)    carve((size_t)NNODES * NH * 4);
  float*     adst   = (float*)    carve((size_t)NNODES * NH * 4);
  unsigned*  mbuf   = (unsigned*) carve((size_t)NNODES * NH * 4);
  float*     sbuf   = (float*)    carve((size_t)NNODES * NH * 4);
  float*     albuf  = (float*)    carve((size_t)ETOT * NH * 4);
  float*     outacc = (float*)    carve((size_t)NNODES * 256 * 4);
  float*     xi     = (float*)    carve((size_t)NNODES * 64 * 4);

  const int* srcI = eidx;
  const int* dstI = eidx + NEDGES;

  auto nblk = [](long long n) { return (int)((n + 255) / 256); };

  // edge encoder + self-loop mean fill
  k_fill_f32<<<1, 256, 0, stream>>>(ewsum, 0.f, 1);
  k_edge_encoder<<<nblk(NEDGES), 256, 0, stream>>>(eattr, ee_w1, ee_b1, ee_w2, ee_b2,
                                                   ew, ewsum, NEDGES);
  k_selfloops<<<nblk(NNODES), 256, 0, stream>>>(ew, ewsum, NEDGES, NNODES);

  const int mtiles = NNODES / 16;  // 50000 = 16*3125, exact

  for (int l = 0; l < 3; ++l) {
    if (l == 0) {
      k_pack_a16<<<nblk((long long)NNODES * 32), 256, 0, stream>>>(x, A16, NNODES, FIN, 32);
      k_pack_bt16<<<nblk(256 * 32), 256, 0, stream>>>(W1, BT16, FIN, 32);
      k_gemm_wmma<32><<<mtiles, 512, 0, stream>>>(A16, BT16, hbuf, NNODES);
    } else {
      k_pack_bt16<<<nblk(256 * 64), 256, 0, stream>>>(Ws + (size_t)(l - 1) * 64 * 256,
                                                      BT16, 64, 64);
      k_gemm_wmma<64><<<mtiles, 512, 0, stream>>>(A16, BT16, hbuf, NNODES);
    }

    k_alpha_nodes<<<nblk((long long)NNODES * NH), 256, 0, stream>>>(
        hbuf, att_s + l * 256, att_d + l * 256, asrc, adst, NNODES);
    k_ce<<<1, 32, 0, stream>>>(We + l * 256, att_e + l * 256, ce);

    k_fill_u32<<<nblk((long long)NNODES * NH), 256, 0, stream>>>(mbuf, NEG_INF_KEY,
                                                                 NNODES * NH);
    k_fill_f32<<<nblk((long long)NNODES * NH), 256, 0, stream>>>(sbuf, 0.f, NNODES * NH);
    k_fill_f32x4<<<nblk((long long)NNODES * 64), 256, 0, stream>>>(
        (float4*)outacc, 0.f, NNODES * 64);

    k_edge_alpha<<<nblk((long long)ETOT * NH), 256, 0, stream>>>(
        srcI, dstI, ew, ce, asrc, adst, albuf, mbuf, ETOT, NEDGES);
    k_edge_softmax<<<nblk((long long)ETOT * NH), 256, 0, stream>>>(
        dstI, albuf, mbuf, sbuf, ETOT, NEDGES);
    k_aggregate<<<(ETOT + 3) / 4, 256, 0, stream>>>(
        srcI, dstI, hbuf, albuf, sbuf, outacc, ETOT, NEDGES);

    k_epilogue<<<nblk((long long)NNODES * 64), 256, 0, stream>>>(
        outacc, bias + l * 64, xi, A16, NNODES, l > 0 ? 1 : 0);
  }

  k_readout<<<NB, 64, 0, stream>>>(xi, batch, ln_g, ln_b, lin1_w, lin1_b,
                                   lin2_w, lin2_b, outp, NNODES);
}
